// _BinarizedWeightConv2D_14345190768698
// MI455X (gfx1250) — matrix operand order repaired and checked by reading, not compiled, not run on hardware
//
#include <hip/hip_runtime.h>

// ---- types matching gfx1250 WMMA operand shapes -------------------------
typedef __attribute__((ext_vector_type(16))) __bf16         v16bf;
typedef __attribute__((ext_vector_type(8)))  float          v8f;
typedef __attribute__((ext_vector_type(8)))  unsigned short u16x8;
typedef __attribute__((ext_vector_type(16))) unsigned short u16x16;

#define CIN   128
#define COUT  256
#define HWDIM 64
#define KTOT  1152   // 9 * 128, K-major order: k = (r*3+s)*128 + c

__device__ __forceinline__ unsigned short f2bf(float f) {
  union { float f; unsigned u; } cv; cv.f = f;
  unsigned u = cv.u;
  unsigned r = u + 0x7FFFu + ((u >> 16) & 1u);   // round-to-nearest-even
  return (unsigned short)(r >> 16);
}

// ---- prep: sign-binarize weights, relayout OIHW -> [O][ (r*3+s)*128 + c ] bf16
__global__ void binarize_weights(const float* __restrict__ w,
                                 unsigned short* __restrict__ wb) {
  int idx = blockIdx.x * 256 + threadIdx.x;
  if (idx >= COUT * KTOT) return;
  int o   = idx / KTOT;
  int rem = idx - o * KTOT;
  int c   = rem / 9;
  int rs  = rem - c * 9;
  float f = w[idx];
  float sgn = (f > 0.f) ? 1.f : ((f < 0.f) ? -1.f : 0.f);
  wb[o * KTOT + rs * CIN + c] = f2bf(sgn);
}

// ---- main: implicit GEMM, software-pipelined.
// Block tile: 128 ochannels (M) x 64 pixels = one (n,h) row (N), K = 1152 in 36 steps of 32.
// 8 waves in a 4(M) x 2(N) grid; each wave owns a 2x2 grid of 16x16 WMMA tiles.
// Pipeline: [store staged regs -> LDS p] barrier [issue global loads k+1] [4x WMMA from p].
__launch_bounds__(256)
__global__ void bconv_wmma(const float* __restrict__ x,
                           const unsigned short* __restrict__ wb,
                           float* __restrict__ out) {
  __shared__ u16x8 Xs[2][64 * 4];    // activations: [pixel w][k 0..31]   (2 x 4KB)
  __shared__ u16x8 Ws[2][128 * 4];   // weights:     [ochan oo][k 0..31]  (2 x 8KB)

  const int tid  = threadIdx.x;
  const int lane = tid & 31;
  const int wave = tid >> 5;

  const int row = blockIdx.x;          // packed (n, h)
  const int n   = row >> 6;
  const int h   = row & 63;
  const int o0  = blockIdx.y * 128;    // output-channel block

  const int m0 = (wave & 3) * 32;      // wave's ochannel sub-block (2 tiles)
  const int n0 = (wave >> 2) * 32;     // wave's pixel sub-block (2 tiles)

  // staging decomposition: thread -> (row 0..63, 8-element k segment 0..3)
  const int st_q = tid & 3;
  const int st_w = tid >> 2;

  const u16x8* wg = (const u16x8*)wb;
  const int wrowA = (o0 + st_w)      * (KTOT / 8);
  const int wrowB = (o0 + st_w + 64) * (KTOT / 8);

  const int half = lane >> 4;
  const int l15  = lane & 15;

  v8f acc[2][2] = {};

  // in-flight staging registers
  float          xr[8];
  u16x8          wr0, wr1;
  unsigned short xm;                   // zero-pad mask for this staged step

  auto stage_load = [&](int step) {
    const int rs = step >> 2, cb = step & 3;           // tap, channel block
    const int r  = rs / 3, s = rs - r * 3;
    const int hi = h + r - 1;
    const bool rowok = (unsigned)hi < (unsigned)HWDIM;
    const int  hic   = rowok ? hi : 0;
    const int  wi    = st_w + s - 1;
    const bool xok   = rowok && ((unsigned)wi < (unsigned)HWDIM);
    const int  wic   = (wi < 0) ? 0 : ((wi > 63) ? 63 : wi);
    xm = xok ? (unsigned short)0xffffu : (unsigned short)0;

    // 8 strided activation loads, clamped addresses, NO conditional use:
    // results always consumed (masked later) so loads batch under one wait.
    const float* xp = x + (((size_t)(n * CIN + cb * 32 + st_q * 8) * HWDIM + hic) * HWDIM + wic);
#pragma unroll
    for (int j = 0; j < 8; ++j)
      xr[j] = xp[(size_t)j * (HWDIM * HWDIM)];

    // two 16B weight rows (K-major bf16)
    const int wk = rs * 16 + cb * 4 + st_q;
    wr0 = wg[wrowA + wk];
    wr1 = wg[wrowB + wk];
  };

  auto stage_store = [&](int p) {
    u16x8 xv;
#pragma unroll
    for (int j = 0; j < 8; ++j)
      xv[j] = (unsigned short)(f2bf(xr[j]) & xm);      // branchless zero-padding
    Xs[p][st_w * 4 + st_q]        = xv;
    Ws[p][st_w * 4 + st_q]        = wr0;
    Ws[p][(st_w + 64) * 4 + st_q] = wr1;
  };

  stage_load(0);
  int p = 0;
  for (int step = 0; step < 36; ++step) {
    stage_store(p);                    // waits on loads issued one step ago
    __syncthreads();                   // buffer p visible; prior compute on p done (k-2 rule)
    if (step < 35) stage_load(step + 1);   // loads fly across this step's WMMAs

    // --- A fragments (weights 16x32): lane<16 -> M=l15, K{0-7,16-23}; lane>=16 -> K{8-15,24-31}
    v16bf a[2], b[2];
#pragma unroll
    for (int mi = 0; mi < 2; ++mi) {
      const int arow = m0 + mi * 16 + l15;
      u16x8 alo = Ws[p][arow * 4 + half];
      u16x8 ahi = Ws[p][arow * 4 + half + 2];
      u16x16 au;
#pragma unroll
      for (int j = 0; j < 8; ++j) { au[j] = alo[j]; au[j + 8] = ahi[j]; }
      a[mi] = __builtin_bit_cast(v16bf, au);
    }
    // --- B fragments (activations 32x16): lane<16 -> N=l15 K{0-7,16-23}; lane>=16 -> K{8-15,24-31}
#pragma unroll
    for (int ni = 0; ni < 2; ++ni) {
      const int brow = n0 + ni * 16 + l15;
      u16x8 blo = Xs[p][brow * 4 + half];
      u16x8 bhi = Xs[p][brow * 4 + half + 2];
      u16x16 bu;
#pragma unroll
      for (int q = 0; q < 8; ++q) { bu[q] = blo[q]; bu[q + 8] = bhi[q]; }
      b[ni] = __builtin_bit_cast(v16bf, bu);
    }
    // --- 2x2 WMMA tiles: each fragment feeds two matmuls
#pragma unroll
    for (int mi = 0; mi < 2; ++mi)
#pragma unroll
      for (int ni = 0; ni < 2; ++ni)
        acc[mi][ni] = __builtin_amdgcn_wmma_f32_16x16x32_bf16(
            false, a[mi], false, b[ni], (short)0, acc[mi][ni], false, false);

    p ^= 1;
  }

  // --- epilogue: D layout puts pixels on lanes -> contiguous w, coalesced stores
#pragma unroll
  for (int mi = 0; mi < 2; ++mi) {
    const int ot = o0 + m0 + mi * 16 + half * 8;
#pragma unroll
    for (int ni = 0; ni < 2; ++ni) {
      const int wt = n0 + ni * 16 + l15;
      float* op = out + (((size_t)(n * COUT + ot) * HWDIM + h) * HWDIM + wt);
#pragma unroll
      for (int v = 0; v < 8; ++v)
        op[(size_t)v * (HWDIM * HWDIM)] = acc[mi][ni][v];
    }
  }
}

extern "C" void kernel_launch(void* const* d_in, const int* in_sizes, int n_in,
                              void* d_out, int out_size, void* d_ws, size_t ws_size,
                              hipStream_t stream) {
  const float* x = (const float*)d_in[0];
  const float* w = (const float*)d_in[1];
  float* out = (float*)d_out;
  unsigned short* wb = (unsigned short*)d_ws;   // 256*1152 bf16 = 576 KB scratch

  binarize_weights<<<(COUT * KTOT + 255) / 256, 256, 0, stream>>>(w, wb);

  dim3 grid(32 * 64 /* (n,h) rows */, COUT / 128 /* ochannel blocks */);
  bconv_wmma<<<grid, 256, 0, stream>>>(x, wb, out);
}
